// DistanceProbeAlternative_61022895341727
// MI455X (gfx1250) — matrix operand order repaired and checked by reading, not compiled, not run on hardware
//
#include <hip/hip_runtime.h>
#include <hip/hip_bf16.h>

#define B_ 32
#define S_ 1024
#define D_ 1024
#define R_ 128

typedef __attribute__((ext_vector_type(16))) __bf16 v16bf;
typedef __attribute__((ext_vector_type(8)))  __bf16 v8bf;
typedef __attribute__((ext_vector_type(8)))  float  v8f;
typedef __attribute__((ext_vector_type(4)))  float  v4f;
typedef __attribute__((ext_vector_type(4)))  unsigned int u32x4;
typedef __attribute__((ext_vector_type(8)))  int i32x8;
typedef __attribute__((ext_vector_type(4)))  int i32x4;

static __device__ __forceinline__ v16bf cat8(v8bf lo, v8bf hi) {
  return __builtin_shufflevector(lo, hi, 0,1,2,3,4,5,6,7,8,9,10,11,12,13,14,15);
}

// Load 8 consecutive fp32 (32B, 16B-aligned) and convert to 8 bf16.
static __device__ __forceinline__ v8bf load_cvt8(const float* __restrict__ p) {
  v4f f0 = *(const v4f*)(p);
  v4f f1 = *(const v4f*)(p + 4);
  v8bf r;
#pragma unroll
  for (int i = 0; i < 4; ++i) { r[i] = (__bf16)f0[i]; r[4 + i] = (__bf16)f1[i]; }
  return r;
}

// ---------------------------------------------------------------------------
// TDM: DMA a 128x128 bf16 tile (row stride R_=128 elems) from global into LDS
// with a 16B pad after every 256B row -> 272B LDS pitch (bank-conflict-free
// column access). D# layout per CDNA5 ISA.
// ---------------------------------------------------------------------------
#define TILE_PITCH 272
#define TILE_BYTES (128 * TILE_PITCH)   // 34816 B per tile

static __device__ __forceinline__ void tdm_load_tile(const __bf16* gaddr,
                                                     unsigned lds_off) {
  unsigned long long ga = (unsigned long long)(size_t)gaddr;
  u32x4 g0;
  g0[0] = 1u;                                   // count=1, user mode
  g0[1] = lds_off;                              // lds_addr (bytes)
  g0[2] = (unsigned)(ga & 0xFFFFFFFFu);         // global_addr[31:0]
  g0[3] = (unsigned)((ga >> 32) & 0x01FFFFFFu)  // global_addr[56:32]
        | (2u << 30);                           // type=2 ("image")
  i32x8 g1;
  g1[0] = (int)((1u << 16)      // data_size = 2 bytes
              | (1u << 20)      // pad_enable
              | (5u << 22)      // pad_interval: every 64 DWORDs (256 B row)
              | (3u << 25));    // pad_amount: 4 DWORDs (16 B)
  g1[1] = (int)(128u << 16);    // tensor_dim0 = 128 (bits 79:48 low half)
  g1[2] = (int)(128u << 16);    // tensor_dim0 hi=0 | tensor_dim1 = 128 (low half)
  g1[3] = (int)(128u << 16);    // tensor_dim1 hi=0 | tile_dim0 = 128
  g1[4] = (int)(128u);          // tile_dim1 = 128 | tile_dim2 = 0
  g1[5] = (int)(128u);          // tensor_dim0_stride[31:0] = 128 elems
  g1[6] = 0;                    // stride0 hi | tensor_dim1_stride lo
  g1[7] = 0;
  i32x4 gz = {0, 0, 0, 0};
#if __clang_major__ >= 23
  i32x8 gz8 = {0, 0, 0, 0, 0, 0, 0, 0};
  __builtin_amdgcn_tensor_load_to_lds(g0, g1, gz, gz, gz8, 0);
#else
  __builtin_amdgcn_tensor_load_to_lds(g0, g1, gz, gz, 0);
#endif
}

// ---------------------------------------------------------------------------
// Stage 1: proj = X @ W^T  (bf16 output), WMMA bf16, W tile staged in LDS
// in per-lane fragment order. 4 waves/WG, each wave: 32 rows x 128 cols.
// ---------------------------------------------------------------------------
__global__ __launch_bounds__(128, 1)
void proj_kernel(const float* __restrict__ X, const float* __restrict__ W,
                 __bf16* __restrict__ P) {
  // fragbuf[c][lane][16] : lane's 16 bf16 B-fragment elements, 32B contiguous
  __shared__ alignas(128) __bf16 fragbuf[8 * 32 * 16];  // 8 KB

  const int t    = threadIdx.x;        // 0..127
  const int lane = t & 31;
  const int wave = t >> 5;
  const int i0   = blockIdx.x * 128 + wave * 32;
  const int l15  = lane & 15;
  const int aoff = (lane & 16) ? 8 : 0;   // A-frag K sub-offset per ISA layout

  v8f acc[2][8];
#pragma unroll
  for (int r = 0; r < 2; ++r)
#pragma unroll
    for (int c = 0; c < 8; ++c)
#pragma unroll
      for (int v = 0; v < 8; ++v) acc[r][c][v] = 0.0f;

  for (int k = 0; k < D_; k += 32) {
    __syncthreads();  // protect previous iteration's fragbuf reads
    {
      // thread t owns W row t: convert W[t][k..k+31] -> fragment-order LDS.
      // B-matrix (32x16) layout: lane n<16 holds col n, K={0..7,16..23}; lane n+16 K={8..15,24..31}.
      const float* wp = W + (size_t)t * D_ + k;
      v8bf w0 = load_cvt8(wp +  0);
      v8bf w1 = load_cvt8(wp +  8);
      v8bf w2 = load_cvt8(wp + 16);
      v8bf w3 = load_cvt8(wp + 24);
      const int c   = t >> 4;
      const int llo = t & 15;
      *(v16bf*)&fragbuf[(c * 32 + llo     ) * 16] = cat8(w0, w2);  // K=k..k+7,k+16..k+23
      *(v16bf*)&fragbuf[(c * 32 + llo + 16) * 16] = cat8(w1, w3);  // K=k+8..k+15,k+24..k+31
    }
    __syncthreads();

    // A fragments from X (fp32 -> bf16). 16-bit A 16x32 layout:
    // lane<16: K {k..k+7, k+16..k+23}; lane>=16: K {k+8..k+15, k+24..k+31}.
    v16bf a[2];
#pragma unroll
    for (int r = 0; r < 2; ++r) {
      const float* xp = X + (size_t)(i0 + 16 * r + l15) * D_ + k + aoff;
      __builtin_prefetch(xp + 32, 0, 0);  // next K step -> global_prefetch_b8
      a[r] = cat8(load_cvt8(xp), load_cvt8(xp + 16));
    }

#pragma unroll
    for (int c = 0; c < 8; ++c) {
      v16bf bf = *(const v16bf*)&fragbuf[(c * 32 + lane) * 16];
      acc[0][c] = __builtin_amdgcn_wmma_f32_16x16x32_bf16(
          false, a[0], false, bf, (short)0, acc[0][c], false, false);
      acc[1][c] = __builtin_amdgcn_wmma_f32_16x16x32_bf16(
          false, a[1], false, bf, (short)0, acc[1][c], false, false);
    }
  }

  // C/D layout: VGPR v -> M = v + 8*(lane>=16), N = lane&15.
  const int hi = lane >> 4;
#pragma unroll
  for (int r = 0; r < 2; ++r)
#pragma unroll
    for (int c = 0; c < 8; ++c)
#pragma unroll
      for (int v = 0; v < 8; ++v) {
        const int m = i0 + 16 * r + 8 * hi + v;
        const int n = 16 * c + l15;
        P[(size_t)m * R_ + n] = (__bf16)acc[r][c][v];
      }
}

// ---------------------------------------------------------------------------
// Stage 1b: per-token squared norms of the bf16 projection.
// ---------------------------------------------------------------------------
__global__ __launch_bounds__(256, 1)
void norms_kernel(const __bf16* __restrict__ P, float* __restrict__ norms) {
  const int row = blockIdx.x * 256 + threadIdx.x;
  const v8bf* p = (const v8bf*)(P + (size_t)row * R_);
  float s = 0.0f;
#pragma unroll
  for (int i = 0; i < R_ / 8; ++i) {
    v8bf x = p[i];
#pragma unroll
    for (int j = 0; j < 8; ++j) { float f = (float)x[j]; s += f * f; }
  }
  norms[row] = s;
}

// ---------------------------------------------------------------------------
// Stage 2: per-batch Gram + distance epilogue. TDM stages the WG's 128-row
// A tile and 128-row B tile (B = proj^T -> same row loads) into padded LDS;
// each wave computes a 64x64 output tile = 4x4 WMMA tiles, K=128 unrolled,
// fragments via conflict-free ds_load_b128 from the 272B-pitch tiles.
// ---------------------------------------------------------------------------
__global__ __launch_bounds__(128, 1)
void dist_kernel(const __bf16* __restrict__ P, const float* __restrict__ norms,
                 float* __restrict__ out) {
  __shared__ alignas(16) unsigned char shbuf[2 * TILE_BYTES];  // ~68 KB

  const int lane = threadIdx.x & 31;
  const int wave = threadIdx.x >> 5;
  const int b    = blockIdx.z;
  const int gti  = blockIdx.y * 128;           // WG row-tile base (global)
  const int gtj  = blockIdx.x * 128;           // WG col-tile base (global)
  const int li   = (wave >> 1) * 64;           // wave's local row base in A tile
  const int lj   = (wave & 1) * 64;            // wave's local row base in B tile
  const int l15  = lane & 15;
  const int hi16 = (lane >> 4) & 1;

  const __bf16* Pb = P + (size_t)b * S_ * R_;
  const float*  nb = norms + (size_t)b * S_;

  // One wave launches both tile DMAs and waits for TENSORcnt; barrier
  // releases the rest. (shbuf is the only __shared__ object -> LDS offset 0.)
  if (wave == 0) {
    tdm_load_tile(Pb + (size_t)gti * R_, 0u);
    tdm_load_tile(Pb + (size_t)gtj * R_, (unsigned)TILE_BYTES);
    __builtin_amdgcn_s_wait_tensorcnt(0);
  }
  // Compiler-level barrier: the TDM builtin is not modeled as writing LDS,
  // so without this AA deletes the ds_loads of shbuf (observed in round 2:
  // dist wmma collapsed 64 -> 4 and ds -> 0). This asm is a potential store.
  asm volatile("" : : "r"(shbuf) : "memory");
  __syncthreads();

  v8f acc[4][4];
#pragma unroll
  for (int r = 0; r < 4; ++r)
#pragma unroll
    for (int c = 0; c < 4; ++c)
#pragma unroll
      for (int v = 0; v < 8; ++v) acc[r][c][v] = 0.0f;

  const unsigned char* abase = shbuf;
  const unsigned char* bbase = shbuf + TILE_BYTES;

#pragma unroll
  for (int k = 0; k < R_; k += 32) {
    const int kc = k >> 3;  // 16B-chunk index of K base within a row
    v16bf a[4], bf[4];
    // A 16x32 layout: lane<16 holds K {k..k+7, k+16..k+23} (chunks kc, kc+2);
    // lane>=16 holds {k+8..k+15, k+24..k+31} (chunks kc+1, kc+3).
#pragma unroll
    for (int r = 0; r < 4; ++r) {
      const int row = li + 16 * r + l15;
      const unsigned char* p = abase + row * TILE_PITCH + (kc + hi16) * 16;
      a[r] = cat8(*(const v8bf*)p, *(const v8bf*)(p + 32));
    }
    // B 32x16 layout: lane<16 holds K {k..k+7, k+16..k+23} (chunks kc, kc+2);
    // lane>=16 holds {k+8..k+15, k+24..k+31} (chunks kc+1, kc+3).
#pragma unroll
    for (int c = 0; c < 4; ++c) {
      const int row = lj + 16 * c + l15;
      const unsigned char* p = bbase + row * TILE_PITCH + (kc + hi16) * 16;
      bf[c] = cat8(*(const v8bf*)p, *(const v8bf*)(p + 32));
    }
#pragma unroll
    for (int r = 0; r < 4; ++r)
#pragma unroll
      for (int c = 0; c < 4; ++c)
        acc[r][c] = __builtin_amdgcn_wmma_f32_16x16x32_bf16(
            false, a[r], false, bf[c], (short)0, acc[r][c], false, false);
  }

  const int ti = gti + li;
  const int tj = gtj + lj;
  float ncol[4];
#pragma unroll
  for (int c = 0; c < 4; ++c) ncol[c] = nb[tj + 16 * c + l15];

  float* ob = out + (size_t)b * S_ * S_;
#pragma unroll
  for (int r = 0; r < 4; ++r) {
    float nrow[8];
#pragma unroll
    for (int v = 0; v < 8; ++v) nrow[v] = nb[ti + 16 * r + 8 * hi16 + v];
#pragma unroll
    for (int c = 0; c < 4; ++c) {
      const int n = tj + 16 * c + l15;
#pragma unroll
      for (int v = 0; v < 8; ++v) {
        const int m = ti + 16 * r + 8 * hi16 + v;
        ob[(size_t)m * S_ + n] = nrow[v] + ncol[c] - 2.0f * acc[r][c][v];
      }
    }
  }
}

// ---------------------------------------------------------------------------
extern "C" void kernel_launch(void* const* d_in, const int* in_sizes, int n_in,
                              void* d_out, int out_size, void* d_ws, size_t ws_size,
                              hipStream_t stream) {
  const float* X = (const float*)d_in[0];   // [B, S, D] fp32
  const float* W = (const float*)d_in[1];   // [R, D]    fp32
  float* out = (float*)d_out;               // [B, S, S] fp32

  __bf16* P     = (__bf16*)d_ws;                                     // 8 MB bf16 proj
  float*  norms = (float*)((char*)d_ws +
                           (size_t)B_ * S_ * R_ * sizeof(__bf16));   // 128 KB

  // Stage 1: projection (bf16), 256 WGs x 128 threads.
  proj_kernel<<<dim3((B_ * S_) / 128), dim3(128), 0, stream>>>(X, W, P);
  // Stage 1b: norms.
  norms_kernel<<<dim3((B_ * S_) / 256), dim3(256), 0, stream>>>(P, norms);
  // Stage 2: Gram + distance, grid (S/128, S/128, B).
  dist_kernel<<<dim3(S_ / 128, S_ / 128, B_), dim3(128), 0, stream>>>(P, norms, out);
}
